// Model_42314017800205
// MI455X (gfx1250) — matrix operand order repaired and checked by reading, not compiled, not run on hardware
//
#include <hip/hip_runtime.h>
#include <cmath>
#include <cstdint>

// ---------------- model constants ----------------
#define BATCHN   4
#define SEQL     1024
#define LBL      384
#define PRD      384
#define LDEC     768
#define ENCIN    7
#define DM       512
#define NH       8
#define DH       64
#define DFF      2048

typedef __attribute__((ext_vector_type(16))) _Float16 v16h;
typedef __attribute__((ext_vector_type(4)))  _Float16 v4h;
typedef __attribute__((ext_vector_type(8)))  float    v8f;
typedef __attribute__((ext_vector_type(4)))  float    v4f;

// ================= WMMA GEMM =================
// C[M,N] = op( A[M,K] @ B[K,N] ), batched via strides (weights shared).
// REQUIRES: M%128==0, N%128==0, K%32==0, lda/ldb/ldc%4==0 (all call sites obey).
// flags: 1=+bias, 2=gelu(exact), 4=accumulate into existing C
#define TBM 128
#define TBN 128
#define TBK 32
#define LSTR 24          // per-lane fragment stride in halves (48B: 16B-aligned, 2-way banks)
#define TSTR (32*LSTR)   // per 16x16-tile stride in halves

__global__ __launch_bounds__(256) void gemm_wmma_kernel(
    const float* __restrict__ A, const float* __restrict__ Bw,
    const float* __restrict__ bias, float* __restrict__ C,
    int M, int N, int K, int lda, int ldb, int ldc,
    long long sA, long long sC, int flags)
{
  A += (long long)blockIdx.z * sA;
  C += (long long)blockIdx.z * sC;
  const int mBase = blockIdx.y * TBM;
  const int nBase = blockIdx.x * TBN;

  // double-buffered fragment-major tiles
  __shared__ alignas(32) _Float16 As[2][8 * TSTR];
  __shared__ alignas(32) _Float16 Bs[2][8 * TSTR];

  const int tid   = threadIdx.x;
  const int lane  = tid & 31;
  const int wid   = tid >> 5;
  const int waveM = wid >> 1;   // 0..3
  const int waveN = wid & 1;    // 0..1

  v8f acc[2][4];
#pragma unroll
  for (int i = 0; i < 2; ++i)
#pragma unroll
    for (int j = 0; j < 4; ++j)
      acc[i][j] = v8f{0.f,0.f,0.f,0.f,0.f,0.f,0.f,0.f};

  // ---- tile staging (vectorized, no guards) ----
  auto stage = [&](int buf, int k0) {
    // A tile: 128x32, each thread 4 x float4 -> 4 x ds_store_b64
#pragma unroll
    for (int i = 0; i < 4; ++i) {
      int idx = tid + i * 256;               // 0..1023 quads
      int ml  = idx >> 3;                    // 0..127
      int kq  = (idx & 7) << 2;              // 0,4,...,28
      v4f v = *reinterpret_cast<const v4f*>(A + (long long)(mBase + ml) * lda + (k0 + kq));
      int mt  = ml >> 4, m = ml & 15;
      int ln  = m + (((kq >> 3) & 1) << 4);
      int jj0 = ((kq >= 16) ? 4 : 0) + ((kq & 7) >> 1);   // even pair index
      v4h h = { (_Float16)v.x, (_Float16)v.y, (_Float16)v.z, (_Float16)v.w };
      *reinterpret_cast<v4h*>(&As[buf][mt * TSTR + ln * LSTR + jj0 * 2]) = h;
    }
    // B tile: 32x128, each thread 4 x float4 -> 16 x ds_store_b16 (lane-scattered)
#pragma unroll
    for (int i = 0; i < 4; ++i) {
      int idx = tid + i * 256;
      int kl  = idx >> 5;                    // 0..31
      int nq  = (idx & 31) << 2;             // 0..124
      v4f v = *reinterpret_cast<const v4f*>(Bw + (long long)(k0 + kl) * ldb + (nBase + nq));
      int nt   = nq >> 4;
      int half = kl & 1;
      int jj   = ((kl & 7) >> 1) + ((kl >= 16) ? 4 : 0);
      int lnb  = (nq & 15) + (((kl >> 3) & 1) << 4);
      _Float16* bp = &Bs[buf][nt * TSTR + jj * 2 + half];
      bp[(lnb + 0) * LSTR] = (_Float16)v.x;
      bp[(lnb + 1) * LSTR] = (_Float16)v.y;
      bp[(lnb + 2) * LSTR] = (_Float16)v.z;
      bp[(lnb + 3) * LSTR] = (_Float16)v.w;
    }
  };

  stage(0, 0);
  __syncthreads();

  int k0 = 0, buf = 0;
  for (;;) {
    int kn = k0 + TBK;
    bool more = kn < K;
    if (more) stage(buf ^ 1, kn);      // prefetch next tile while WMMAs run

    v16h af[2], bf[4];
#pragma unroll
    for (int i = 0; i < 2; ++i)
      af[i] = *reinterpret_cast<const v16h*>(&As[buf][(waveM * 2 + i) * TSTR + lane * LSTR]);
#pragma unroll
    for (int j = 0; j < 4; ++j)
      bf[j] = *reinterpret_cast<const v16h*>(&Bs[buf][(waveN * 4 + j) * TSTR + lane * LSTR]);
#pragma unroll
    for (int i = 0; i < 2; ++i)
#pragma unroll
      for (int j = 0; j < 4; ++j)
        acc[i][j] = __builtin_amdgcn_wmma_f32_16x16x32_f16(
            false, af[i], false, bf[j], (short)0, acc[i][j], false, false);

    if (!more) break;
    k0 = kn; buf ^= 1;
    __syncthreads();
  }

  // ---- epilogue (dims tile-aligned; no guards) ----
#pragma unroll
  for (int i = 0; i < 2; ++i) {
#pragma unroll
    for (int j = 0; j < 4; ++j) {
      int n  = nBase + (waveN * 4 + j) * 16 + (lane & 15);
      int mb = mBase + (waveM * 2 + i) * 16 + ((lane >> 4) ? 8 : 0);
      float bv = (flags & 1) ? bias[n] : 0.f;
#pragma unroll
      for (int r = 0; r < 8; ++r) {
        int m = mb + r;
        float v = acc[i][j][r] + bv;
        if (flags & 4) v += C[(long long)m * ldc + n];
        if (flags & 2) v = 0.5f * v * (1.f + erff(v * 0.70710678118f));
        C[(long long)m * ldc + n] = v;
      }
    }
  }
}

// ================= auxiliary kernels =================

__device__ __forceinline__ unsigned mix_hash(unsigned a) {
  a ^= a >> 16; a *= 0x7feb352dU; a ^= a >> 15; a *= 0x846ca68bU; a ^= a >> 16;
  return a;
}

// data_embedding: circular conv(3, Cin->512) + positional + x_mark @ temp_w
__global__ void embed_kernel(const float* __restrict__ x, const float* __restrict__ xm,
                             const float* __restrict__ cw, const float* __restrict__ tw,
                             float* __restrict__ out, int L, int Cin)
{
  long long gid = (long long)blockIdx.x * blockDim.x + threadIdx.x;
  long long total = (long long)BATCHN * L * DM;
  if (gid >= total) return;
  int d = (int)(gid % DM);
  int t = (int)((gid / DM) % L);
  int b = (int)(gid / ((long long)DM * L));
  float acc = 0.f;
  for (int k = 0; k < 3; ++k) {
    int ts = t - 1 + k; if (ts < 0) ts += L; if (ts >= L) ts -= L;
    const float* xr = x + ((long long)(b * L + ts)) * Cin;
    const float* wr = cw + ((long long)k * Cin) * DM + d;
    for (int c = 0; c < Cin; ++c) acc += xr[c] * wr[(long long)c * DM];
  }
  const float* xmr = xm + ((long long)(b * L + t)) * 4;
  for (int f = 0; f < 4; ++f) acc += xmr[f] * tw[f * DM + d];
  int i2 = d & ~1;
  float freq = __expf(-(float)i2 * (9.210340371976184f / (float)DM));
  float ang  = (float)t * freq;
  acc += (d & 1) ? cosf(ang) : sinf(ang);
  out[gid] = acc;
}

__global__ void decinp_kernel(const float* __restrict__ bx, float* __restrict__ din)
{
  int gid = blockIdx.x * blockDim.x + threadIdx.x;
  int total = BATCHN * LDEC * ENCIN;
  if (gid >= total) return;
  int c = gid % ENCIN;
  int t = (gid / ENCIN) % LDEC;
  int b = gid / (ENCIN * LDEC);
  din[gid] = (t < LBL) ? bx[((long long)(b * SEQL + (SEQL - LBL) + t)) * ENCIN + c] : 0.f;
}

// layernorm over 512 cols; optional residual; in-place safe per row
__global__ __launch_bounds__(256) void ln_kernel(
    const float* __restrict__ x, const float* __restrict__ res,
    const float* __restrict__ g, const float* __restrict__ bta,
    float* __restrict__ y, int rows)
{
  int row = blockIdx.x;
  if (row >= rows) return;
  __shared__ float red[256];
  int tid = threadIdx.x;
  const float* xr = x + (long long)row * DM;
  float v0 = xr[tid], v1 = xr[tid + 256];
  if (res) { const float* rr = res + (long long)row * DM; v0 += rr[tid]; v1 += rr[tid + 256]; }
  red[tid] = v0 + v1; __syncthreads();
  for (int o = 128; o > 0; o >>= 1) { if (tid < o) red[tid] += red[tid + o]; __syncthreads(); }
  float mu = red[0] / (float)DM; __syncthreads();
  float d0 = v0 - mu, d1 = v1 - mu;
  red[tid] = d0 * d0 + d1 * d1; __syncthreads();
  for (int o = 128; o > 0; o >>= 1) { if (tid < o) red[tid] += red[tid + o]; __syncthreads(); }
  float inv = rsqrtf(red[0] / (float)DM + 1e-5f);
  float* yr = y + (long long)row * DM;
  yr[tid]       = d0 * inv * g[tid]       + bta[tid];
  yr[tid + 256] = d1 * inv * g[tid + 256] + bta[tid + 256];
}

// sparsity measure M = max_u(qk_s) - sum_u(qk_s)/Lk over hashed K samples
__global__ void meas_kernel(const float* __restrict__ Q, const float* __restrict__ Kb,
                            float* __restrict__ Mm, int Lq, int Lk, int Up, unsigned seed)
{
  int gid = blockIdx.x * blockDim.x + threadIdx.x;
  int total = BATCHN * NH * Lq;
  if (gid >= total) return;
  int l = gid % Lq, bh = gid / Lq;
  int h = bh % NH, b = bh / NH;
  const float* q = Q + ((long long)(b * Lq + l)) * DM + h * DH;
  float mx = -3.4e38f, sm = 0.f;
  for (int u = 0; u < Up; ++u) {
    unsigned r = mix_hash((unsigned)l * 0x9E3779B9u ^ ((unsigned)u * 0x85EBCA6Bu) ^ (seed * 0xC2B2AE35u)) % (unsigned)Lk;
    const float* kr = Kb + ((long long)(b * Lk + (int)r)) * DM + h * DH;
    float d = 0.f;
#pragma unroll 8
    for (int t = 0; t < DH; ++t) d += q[t] * kr[t];
    mx = fmaxf(mx, d); sm += d;
  }
  Mm[gid] = mx - sm / (float)Lk;
}

// destructive top-u selection per (b,h)
__global__ void topk_kernel(float* __restrict__ Mm, int* __restrict__ idx, int Lq, int u)
{
  int bh = threadIdx.x;
  if (bh >= BATCHN * NH) return;
  float* row = Mm + (long long)bh * Lq;
  int* out = idx + bh * u;
  for (int s = 0; s < u; ++s) {
    float best = -3.4e38f; int bj = 0;
    for (int j = 0; j < Lq; ++j) { float v = row[j]; if (v > best) { best = v; bj = j; } }
    out[s] = bj; row[bj] = -3.4e38f;
  }
}

// base context: mean(v) broadcast (non-causal) or cumsum(v) (causal)
__global__ void ctx_fill_kernel(const float* __restrict__ V, float* __restrict__ ctx,
                                int Lq, int Lk, int mask)
{
  int gid = blockIdx.x * blockDim.x + threadIdx.x;
  if (gid >= BATCHN * NH * DH) return;
  int d = gid % DH, bh = gid / DH;
  int h = bh % NH, b = bh / NH;
  if (mask) {
    float run = 0.f;
    for (int l = 0; l < Lq; ++l) {
      run += V[((long long)(b * Lk + l)) * DM + h * DH + d];
      ctx[((long long)(b * Lq + l)) * DM + h * DH + d] = run;
    }
  } else {
    float s = 0.f;
    for (int l = 0; l < Lk; ++l) s += V[((long long)(b * Lk + l)) * DM + h * DH + d];
    float mean = s / (float)Lk;
    for (int l = 0; l < Lq; ++l)
      ctx[((long long)(b * Lq + l)) * DM + h * DH + d] = mean;
  }
}

// per selected query row: scores -> softmax -> update, scatter into ctx
__global__ __launch_bounds__(256) void attn_upd_kernel(
    const float* __restrict__ Q, const float* __restrict__ Kb, const float* __restrict__ V,
    const int* __restrict__ idx, float* __restrict__ ctx, int Lq, int Lk, int u, int mask)
{
  __shared__ float sc[1024];
  __shared__ float red[256];
  __shared__ float qr[DH];
  __shared__ float part[4][DH];
  int lin = blockIdx.x;
  int ui = lin % u, bh = lin / u;
  int h = bh % NH, b = bh / NH;
  int tid = threadIdx.x;
  int row = idx[bh * u + ui];
  if (tid < DH) qr[tid] = Q[((long long)(b * Lq + row)) * DM + h * DH + tid];
  __syncthreads();
  float lm = -3.4e38f;
  for (int j = tid; j < Lk; j += 256) {
    const float* kr = Kb + ((long long)(b * Lk + j)) * DM + h * DH;
    float d = 0.f;
#pragma unroll 8
    for (int t = 0; t < DH; ++t) d += qr[t] * kr[t];
    d *= 0.125f;                         // 64^-0.5
    if (mask && j > row) d = -3.4e38f;
    sc[j] = d;
    lm = fmaxf(lm, d);
  }
  red[tid] = lm; __syncthreads();
  for (int o = 128; o > 0; o >>= 1) { if (tid < o) red[tid] = fmaxf(red[tid], red[tid + o]); __syncthreads(); }
  float mval = red[0]; __syncthreads();
  float ls = 0.f;
  for (int j = tid; j < Lk; j += 256) { float e = expf(sc[j] - mval); sc[j] = e; ls += e; }
  red[tid] = ls; __syncthreads();
  for (int o = 128; o > 0; o >>= 1) { if (tid < o) red[tid] += red[tid + o]; __syncthreads(); }
  float denom = red[0]; __syncthreads();
  int d = tid & 63, grp = tid >> 6;
  float a = 0.f;
  for (int j = grp; j < Lk; j += 4)
    a += sc[j] * V[((long long)(b * Lk + j)) * DM + h * DH + d];
  part[grp][d] = a; __syncthreads();
  if (tid < DH) {
    float s = (part[0][tid] + part[1][tid] + part[2][tid] + part[3][tid]) / denom;
    ctx[((long long)(b * Lq + row)) * DM + h * DH + tid] = s;
  }
}

__global__ void pad_kernel(const float* __restrict__ x, float* __restrict__ xp, int L)
{
  long long gid = (long long)blockIdx.x * blockDim.x + threadIdx.x;
  long long total = (long long)BATCHN * (L + 2) * DM;
  if (gid >= total) return;
  int d = (int)(gid % DM);
  int t = (int)((gid / DM) % (L + 2));
  int b = (int)(gid / ((long long)DM * (L + 2)));
  int src = (t == 0) ? (L - 1) : ((t == L + 1) ? 0 : (t - 1));
  xp[gid] = x[((long long)(b * L + src)) * DM + d];
}

__global__ void bn_elu_kernel(float* __restrict__ y, const float* __restrict__ mean,
                              const float* __restrict__ var, const float* __restrict__ g,
                              const float* __restrict__ bta, long long total)
{
  long long gid = (long long)blockIdx.x * blockDim.x + threadIdx.x;
  if (gid >= total) return;
  int d = (int)(gid % DM);
  float v = (y[gid] - mean[d]) * rsqrtf(var[d] + 1e-5f) * g[d] + bta[d];
  y[gid] = (v > 0.f) ? v : expm1f(v);
}

__global__ void maxpool_kernel(const float* __restrict__ x, float* __restrict__ y, int L)
{
  int Lo = L / 2;
  long long gid = (long long)blockIdx.x * blockDim.x + threadIdx.x;
  long long total = (long long)BATCHN * Lo * DM;
  if (gid >= total) return;
  int d = (int)(gid % DM);
  int t = (int)((gid / DM) % Lo);
  int b = (int)(gid / ((long long)DM * Lo));
  float m = -3.4e38f;
  for (int dj = -1; dj <= 1; ++dj) {
    int j = 2 * t + dj;
    if (j >= 0 && j < L) m = fmaxf(m, x[((long long)(b * L + j)) * DM + d]);
  }
  y[gid] = m;
}

__global__ void proj_kernel(const float* __restrict__ dec, const float* __restrict__ pw,
                            const float* __restrict__ pb, float* __restrict__ out)
{
  int gid = blockIdx.x * blockDim.x + threadIdx.x;
  int total = BATCHN * PRD * ENCIN;
  if (gid >= total) return;
  int c = gid % ENCIN;
  int t = (gid / ENCIN) % PRD;
  int b = gid / (ENCIN * PRD);
  const float* xr = dec + ((long long)(b * LDEC + LBL + t)) * DM;
  float a = pb[c];
#pragma unroll 8
  for (int k = 0; k < DM; ++k) a += xr[k] * pw[k * ENCIN + c];
  out[gid] = a;
}

// ================= host side =================

struct AttnW { const float *bk,*bo,*bq,*bv,*wk,*wo,*wq,*wv; };
struct FfnW  { const float *b1,*b2,*w1,*w2; };
struct LnW   { const float *b,*g; };
struct ConvW { const float *b,*bn_b,*bn_g,*bn_mean,*bn_var,*w; };
struct EmbW  { const float *conv_w,*temp_w; };

static inline void gemm(hipStream_t s, const float* A, const float* B, const float* bias,
                        float* C, int M, int N, int K, int lda, int ldb, int ldc,
                        long long sA, long long sC, int batches, int flags)
{
  dim3 g((N + TBN - 1) / TBN, (M + TBM - 1) / TBM, batches);
  gemm_wmma_kernel<<<g, dim3(256), 0, s>>>(A, B, bias, C, M, N, K, lda, ldb, ldc, sA, sC, flags);
}

static inline int sparse_cnt(int L)
{
  int v = (int)(5.0 * std::ceil(std::log((double)L)));
  return v < L ? v : L;
}

static void run_attention(hipStream_t s, const float* xq, int Lq, const float* xkv, int Lk,
                          const AttnW& w, int mask, unsigned seed,
                          float* Qb, float* Kb, float* Vb, float* CT,
                          float* Mm, int* IDX, float* dst)
{
  gemm(s, xq,  w.wq, w.bq, Qb, BATCHN * Lq, DM, DM, DM, DM, DM, 0, 0, 1, 1);
  gemm(s, xkv, w.wk, w.bk, Kb, BATCHN * Lk, DM, DM, DM, DM, DM, 0, 0, 1, 1);
  gemm(s, xkv, w.wv, w.bv, Vb, BATCHN * Lk, DM, DM, DM, DM, DM, 0, 0, 1, 1);
  int Up = sparse_cnt(Lk), u = sparse_cnt(Lq);
  int nq = BATCHN * NH * Lq;
  meas_kernel<<<(nq + 255) / 256, 256, 0, s>>>(Qb, Kb, Mm, Lq, Lk, Up, seed);
  topk_kernel<<<1, 32, 0, s>>>(Mm, IDX, Lq, u);
  ctx_fill_kernel<<<(BATCHN * NH * DH + 255) / 256, 256, 0, s>>>(Vb, CT, Lq, Lk, mask);
  attn_upd_kernel<<<dim3(BATCHN * NH * u), 256, 0, s>>>(Qb, Kb, Vb, IDX, CT, Lq, Lk, u, mask);
  gemm(s, CT, w.wo, w.bo, dst, BATCHN * Lq, DM, DM, DM, DM, DM, 0, 0, 1, 1);
}

extern "C" void kernel_launch(void* const* d_in, const int* in_sizes, int n_in,
                              void* d_out, int out_size, void* d_ws, size_t ws_size,
                              hipStream_t stream)
{
  (void)in_sizes; (void)n_in; (void)out_size; (void)ws_size;
  size_t cur = 0;
  auto nx = [&]() { return (const float*)d_in[cur++]; };

  // jax tree_flatten order: dict keys alphabetical, lists in order
  const float* batch_x      = nx();
  const float* batch_x_mark = nx();
  const float* batch_y_mark = nx();

  ConvW conv[2];
  for (int i = 0; i < 2; ++i) { conv[i] = { nx(), nx(), nx(), nx(), nx(), nx() }; }
  EmbW dec_emb = { nx(), nx() };
  struct DecL { AttnW cross; FfnW ffn; LnW ln1, ln2, ln3; AttnW self; } decl[2];
  for (int i = 0; i < 2; ++i) {
    decl[i].cross = { nx(), nx(), nx(), nx(), nx(), nx(), nx(), nx() };
    decl[i].ffn   = { nx(), nx(), nx(), nx() };
    decl[i].ln1   = { nx(), nx() };
    decl[i].ln2   = { nx(), nx() };
    decl[i].ln3   = { nx(), nx() };
    decl[i].self  = { nx(), nx(), nx(), nx(), nx(), nx(), nx(), nx() };
  }
  LnW dec_norm = { nx(), nx() };
  EmbW enc_emb = { nx(), nx() };
  struct EncL { AttnW attn; FfnW ffn; LnW ln1, ln2; } encl[3];
  for (int i = 0; i < 3; ++i) {
    encl[i].attn = { nx(), nx(), nx(), nx(), nx(), nx(), nx(), nx() };
    encl[i].ffn  = { nx(), nx(), nx(), nx() };
    encl[i].ln1  = { nx(), nx() };
    encl[i].ln2  = { nx(), nx() };
  }
  LnW enc_norm = { nx(), nx() };
  const float* proj_b = nx();
  const float* proj_w = nx();

  // ---- workspace bump allocation (floats) ----
  float* ws = (float*)d_ws;
  size_t off = 0;
  auto alloc = [&](size_t n) { float* p = ws + off; off += n; return p; };
  const size_t ACT = (size_t)BATCHN * SEQL * DM;       // 2M floats
  float* X   = alloc(ACT);                             // encoder stream
  float* Qb  = alloc(ACT);
  float* Kb  = alloc(ACT);
  float* Vb  = alloc(ACT);
  float* CT  = alloc(ACT);
  float* T4  = alloc(ACT);
  float* Hf  = alloc((size_t)BATCHN * SEQL * DFF);     // FFN hidden
  float* XP  = alloc((size_t)BATCHN * (SEQL + 2) * DM);
  float* Yc  = alloc(ACT);                             // conv out
  float* DEC = alloc((size_t)BATCHN * LDEC * DM);
  float* Mm  = alloc((size_t)BATCHN * NH * SEQL);
  int*   IDX = (int*)alloc((size_t)BATCHN * NH * 64);
  float* DIN = alloc((size_t)BATCHN * LDEC * ENCIN);

  // ================= encoder =================
  int L = SEQL;
  {
    long long n = (long long)BATCHN * L * DM;
    embed_kernel<<<(int)((n + 255) / 256), 256, 0, stream>>>(
        batch_x, batch_x_mark, enc_emb.conv_w, enc_emb.temp_w, X, L, ENCIN);
  }
  for (int i = 0; i < 3; ++i) {
    // self attention (non-causal prob-sparse)
    run_attention(stream, X, L, X, L, encl[i].attn, 0, 1000u + i, Qb, Kb, Vb, CT, Mm, IDX, Qb);
    ln_kernel<<<BATCHN * L, 256, 0, stream>>>(X, Qb, encl[i].ln1.g, encl[i].ln1.b, X, BATCHN * L);
    // FFN
    gemm(stream, X,  encl[i].ffn.w1, encl[i].ffn.b1, Hf, BATCHN * L, DFF, DM, DM, DFF, DFF, 0, 0, 1, 1 | 2);
    gemm(stream, Hf, encl[i].ffn.w2, encl[i].ffn.b2, T4, BATCHN * L, DM, DFF, DFF, DM, DM, 0, 0, 1, 1);
    ln_kernel<<<BATCHN * L, 256, 0, stream>>>(X, T4, encl[i].ln2.g, encl[i].ln2.b, X, BATCHN * L);
    // distil conv
    if (i < 2) {
      long long n = (long long)BATCHN * (L + 2) * DM;
      pad_kernel<<<(int)((n + 255) / 256), 256, 0, stream>>>(X, XP, L);
      for (int k = 0; k < 3; ++k) {
        gemm(stream, XP + k * DM, conv[i].w + (size_t)k * DM * DM, conv[i].b, Yc,
             L, DM, DM, DM, DM, DM,
             (long long)(L + 2) * DM, (long long)L * DM, BATCHN, (k == 0) ? 1 : 4);
      }
      long long ne = (long long)BATCHN * L * DM;
      bn_elu_kernel<<<(int)((ne + 255) / 256), 256, 0, stream>>>(
          Yc, conv[i].bn_mean, conv[i].bn_var, conv[i].bn_g, conv[i].bn_b, ne);
      long long no = (long long)BATCHN * (L / 2) * DM;
      maxpool_kernel<<<(int)((no + 255) / 256), 256, 0, stream>>>(Yc, X, L);
      L /= 2;
    }
  }
  ln_kernel<<<BATCHN * L, 256, 0, stream>>>(X, nullptr, enc_norm.g, enc_norm.b, X, BATCHN * L);
  const int Lenc = L;   // 256

  // ================= decoder =================
  decinp_kernel<<<(BATCHN * LDEC * ENCIN + 255) / 256, 256, 0, stream>>>(batch_x, DIN);
  {
    long long n = (long long)BATCHN * LDEC * DM;
    embed_kernel<<<(int)((n + 255) / 256), 256, 0, stream>>>(
        DIN, batch_y_mark, dec_emb.conv_w, dec_emb.temp_w, DEC, LDEC, ENCIN);
  }
  for (int i = 0; i < 2; ++i) {
    run_attention(stream, DEC, LDEC, DEC, LDEC, decl[i].self, 1, 2000u + i, Qb, Kb, Vb, CT, Mm, IDX, Qb);
    ln_kernel<<<BATCHN * LDEC, 256, 0, stream>>>(DEC, Qb, decl[i].ln1.g, decl[i].ln1.b, DEC, BATCHN * LDEC);
    run_attention(stream, DEC, LDEC, X, Lenc, decl[i].cross, 0, 3000u + i, Qb, Kb, Vb, CT, Mm, IDX, Qb);
    ln_kernel<<<BATCHN * LDEC, 256, 0, stream>>>(DEC, Qb, decl[i].ln2.g, decl[i].ln2.b, DEC, BATCHN * LDEC);
    gemm(stream, DEC, decl[i].ffn.w1, decl[i].ffn.b1, Hf, BATCHN * LDEC, DFF, DM, DM, DFF, DFF, 0, 0, 1, 1 | 2);
    gemm(stream, Hf,  decl[i].ffn.w2, decl[i].ffn.b2, T4, BATCHN * LDEC, DM, DFF, DFF, DM, DM, 0, 0, 1, 1);
    ln_kernel<<<BATCHN * LDEC, 256, 0, stream>>>(DEC, T4, decl[i].ln3.g, decl[i].ln3.b, DEC, BATCHN * LDEC);
  }
  ln_kernel<<<BATCHN * LDEC, 256, 0, stream>>>(DEC, nullptr, dec_norm.g, dec_norm.b, DEC, BATCHN * LDEC);

  proj_kernel<<<(BATCHN * PRD * ENCIN + 255) / 256, 256, 0, stream>>>(DEC, proj_w, proj_b, (float*)d_out);
}
